// GraphEmbedding_58643483459777
// MI455X (gfx1250) — matrix operand order repaired and checked by reading, not compiled, not run on hardware
//
#include <hip/hip_runtime.h>
#include <math.h>

// ---------------------------------------------------------------------------
// TGN graph-embedding for MI455X (gfx1250, wave32, WMMA).
//
// out[b] = (nf[src]+Σnf[nbr]/cnt) @ Wn  +  (Σef[eid]/cnt) @ We
//        + (cos(b_vec)+Σcos(dt*w+b)/cnt) @ Wt  +  mem[src]+Σmem[nbr]/cnt
//
// All matmul operands folded over K=20 first (linearity of masked mean),
// then one f16 WMMA GEMM:  [4096 x 512] x [512 x 128] -> f32.
// ---------------------------------------------------------------------------

typedef __attribute__((ext_vector_type(16))) _Float16 v16h;
typedef __attribute__((ext_vector_type(8)))  _Float16 v8h;
typedef __attribute__((ext_vector_type(8)))  float    v8f;

#define TGN_B    4096
#define TGN_K    20
#define TGN_DN   172   // node/edge feature dim
#define TGN_DT   100   // time encoding dim
#define TGN_DE   128   // embedding dim
#define TGN_KPAD 512   // 192 (node) + 192 (edge) + 128 (time), each mult. of 32

// ---------------------------------------------------------------------------
// Kernel 1: pack [W_node ; W_edge ; W_time] (rows zero-padded to 512) into f16
// in WMMA B-fragment order:  Wb[(s*128 + n)*32 + kl] = W[32*s + kl][n]
// so a lane's 16 halves (K = 32s + hi*16 + 0..15, fixed N) are contiguous.
// ---------------------------------------------------------------------------
__global__ __launch_bounds__(256) void tgn_pack_w(
    const float* __restrict__ Wn, const float* __restrict__ We,
    const float* __restrict__ Wt, _Float16* __restrict__ Wb) {
  int i = blockIdx.x * 256 + threadIdx.x;
  if (i >= TGN_KPAD * TGN_DE) return;
  int kl = i & 31;          // K within 32-wide k-step
  int n  = (i >> 5) & 127;  // output column
  int s  = i >> 12;         // k-step index (0..15)
  int r  = s * 32 + kl;     // concatenated K row (0..511)
  float v = 0.0f;
  if (r < 172)                    v = Wn[r * TGN_DE + n];
  else if (r >= 192 && r < 364)   v = We[(r - 192) * TGN_DE + n];
  else if (r >= 384 && r < 484)   v = Wt[(r - 384) * TGN_DE + n];
  Wb[i] = (_Float16)v;
}

// ---------------------------------------------------------------------------
// Kernel 2: per-source gather + masked-mean fold (the bandwidth-bound phase).
// One block per source b. Produces:
//   Acat[b, 0:512]  (f16):  [node 0..171 | 0 | edge 0..171 | 0 | time 0..99 | 0]
//   Mmem[b, 0:128]  (f32):  mem[src] + Σ_valid mem[nbr] / cnt
// ---------------------------------------------------------------------------
__global__ __launch_bounds__(256) void tgn_gather(
    const float* __restrict__ nf,  const float* __restrict__ ef,
    const float* __restrict__ mem, const float* __restrict__ ts,
    const float* __restrict__ et,  const float* __restrict__ tw,
    const float* __restrict__ tb,  const int* __restrict__ src,
    const int* __restrict__ nbr,   const int* __restrict__ eid,
    _Float16* __restrict__ Acat,   float* __restrict__ Mmem) {
  const int b = blockIdx.x;
  const int t = threadIdx.x;

  __shared__ int   s_nbr[TGN_K];
  __shared__ int   s_eid[TGN_K];
  __shared__ float s_et [TGN_K];
  __shared__ float s_ts;
  __shared__ int   s_src;

  if (t < TGN_K) {
    s_nbr[t] = nbr[b * TGN_K + t];
    s_eid[t] = eid[b * TGN_K + t];
    s_et [t] = et [b * TGN_K + t];
  }
  if (t == 0) { s_ts = ts[b]; s_src = src[b]; }
  __syncthreads();

  int cnt = 0;
#pragma unroll
  for (int k = 0; k < TGN_K; ++k) cnt += (s_nbr[k] != 0);
  const float inv    = 1.0f / (float)(cnt < 1 ? 1 : cnt);
  const float tstamp = s_ts;
  const int   srcid  = s_src;

  // Each thread owns columns t and t+256 of the 512-wide concatenated row.
  for (int c = t; c < TGN_KPAD; c += 256) {
    float val = 0.0f;
    if (c < 172) {                                  // node-feature segment
      float acc = 0.0f;
      for (int k = 0; k < TGN_K; ++k) {
        int nn = s_nbr[k];
        if (nn != 0) acc += nf[(size_t)nn * TGN_DN + c];
      }
      val = nf[(size_t)srcid * TGN_DN + c] + acc * inv;
    } else if (c >= 192 && c < 364) {               // edge-feature segment
      int d = c - 192;
      float acc = 0.0f;
      for (int k = 0; k < TGN_K; ++k) {
        if (s_nbr[k] != 0) acc += ef[(size_t)s_eid[k] * TGN_DN + d];
      }
      val = acc * inv;
    } else if (c >= 384 && c < 484) {               // time-encoding segment
      int d = c - 384;
      float w  = tw[d];
      float bb = tb[d];
      float acc = 0.0f;
      for (int k = 0; k < TGN_K; ++k) {
        if (s_nbr[k] != 0) acc += cosf((tstamp - s_et[k]) * w + bb);
      }
      val = cosf(bb) + acc * inv;                   // src time emb: t=0 -> cos(b)
    }
    Acat[(size_t)b * TGN_KPAD + c] = (_Float16)val;
  }

  // Memory path stays f32 end-to-end (bypasses the matrix unit).
  if (t < TGN_DE) {
    float acc = 0.0f;
    for (int k = 0; k < TGN_K; ++k) {
      int nn = s_nbr[k];
      if (nn != 0) acc += mem[(size_t)nn * TGN_DE + t];
    }
    Mmem[(size_t)b * TGN_DE + t] = mem[(size_t)srcid * TGN_DE + t] + acc * inv;
  }
}

// ---------------------------------------------------------------------------
// Kernel 3: WMMA GEMM  out[4096,128] = Acat[4096,512] @ Wb[512,128] + Mmem.
// 256 blocks (M tiles of 16), 8 waves/block, wave w owns N-tile w.
// 16 k-steps of v_wmma_f32_16x16x32_f16, f32 accumulation.
//
// A fragment (contiguous K order, same as B packing): lane M = lane&15;
//   lanes 0-15 halves = K{0..15}, lanes 16-31 = K{16..31}
//   -> two aligned 16-byte loads per step.
// B fragment: packed contiguously by tgn_pack_w -> one aligned 32-byte load.
// C/D: VGPR r holds M = r (lanes 0-15) / M = 8+r (lanes 16-31), N = lane&15.
// ---------------------------------------------------------------------------
__global__ __launch_bounds__(256) void tgn_gemm(
    const _Float16* __restrict__ Acat, const _Float16* __restrict__ Wb,
    const float* __restrict__ Mmem, float* __restrict__ out) {
  const int wave  = threadIdx.x >> 5;
  const int lane  = threadIdx.x & 31;
  const int laneM = lane & 15;
  const int hi    = lane >> 4;          // 0: low K half, 1: high K half
  const int m0    = blockIdx.x * 16;
  const int n     = wave * 16 + laneM;  // this lane's output column

  const _Float16* Arow = Acat + (size_t)(m0 + laneM) * TGN_KPAD;

  v8f c = {};
#pragma unroll
  for (int s = 0; s < TGN_KPAD / 32; ++s) {
    const int offA = s * 32 + hi * 16;
    union { v16h v; struct { v8h lo; v8h hi; } p; } ua;
    ua.p.lo = *(const v8h*)(Arow + offA);        // K = 32s + hi*16 + 0..7
    ua.p.hi = *(const v8h*)(Arow + offA + 8);    // K = 32s + hi*16 + 8..15
    const v16h bfrag =
        *(const v16h*)(Wb + ((size_t)(s * TGN_DE + n) * 32) + hi * 16);
    c = __builtin_amdgcn_wmma_f32_16x16x32_f16(
        false, ua.v, false, bfrag, (short)0, c, false, false);
  }

#pragma unroll
  for (int r = 0; r < 8; ++r) {
    const int row = m0 + hi * 8 + r;
    const size_t o = (size_t)row * TGN_DE + n;
    out[o] = c[r] + Mmem[o];
  }
}

// ---------------------------------------------------------------------------
extern "C" void kernel_launch(void* const* d_in, const int* in_sizes, int n_in,
                              void* d_out, int out_size, void* d_ws, size_t ws_size,
                              hipStream_t stream) {
  const float* nf  = (const float*)d_in[0];   // node_features [500000,172]
  const float* ef  = (const float*)d_in[1];   // edge_features [1000000,172]
  const float* mem = (const float*)d_in[2];   // memory        [500000,128]
  const float* ts  = (const float*)d_in[3];   // timestamps    [4096]
  const float* et  = (const float*)d_in[4];   // edge_times    [4096,20]
  const float* tw  = (const float*)d_in[5];   // time_w        [100]
  const float* tb  = (const float*)d_in[6];   // time_b        [100]
  const float* Wn  = (const float*)d_in[7];   // W_node        [172,128]
  const float* We  = (const float*)d_in[8];   // W_edge        [172,128]
  const float* Wt  = (const float*)d_in[9];   // W_time        [100,128]
  const int* src = (const int*)d_in[10];      // source_nodes  [4096]
  const int* nbr = (const int*)d_in[11];      // neighbors     [4096,20]
  const int* eid = (const int*)d_in[12];      // edge_idxs     [4096,20]
  float* out = (float*)d_out;

  char* ws = (char*)d_ws;
  _Float16* Acat = (_Float16*)ws;                                // 4 MiB
  _Float16* Wb   = (_Float16*)(ws + (size_t)4 * 1024 * 1024);    // 128 KiB
  float*    Mm   = (float*)   (ws + (size_t)4 * 1024 * 1024
                                  + (size_t)256 * 1024);         // 2 MiB

  tgn_pack_w<<<(TGN_KPAD * TGN_DE + 255) / 256, 256, 0, stream>>>(Wn, We, Wt, Wb);
  tgn_gather<<<TGN_B, 256, 0, stream>>>(nf, ef, mem, ts, et, tw, tb,
                                        src, nbr, eid, Acat, Mm);
  tgn_gemm<<<TGN_B / 16, 256, 0, stream>>>(Acat, Wb, Mm, out);
}
